// RE_12979391169264
// MI455X (gfx1250) — matrix operand order repaired and checked by reading, not compiled, not run on hardware
//
#include <hip/hip_runtime.h>
#include <hip/hip_bf16.h>

// ---------------------------------------------------------------------------
// Problem constants (reference: B=32, M=64, NV=65, H=960, D=480, ROUNDS=5)
// ---------------------------------------------------------------------------
constexpr int Bn = 32;
constexpr int Mn = 64;
constexpr int NVn = 65;
constexpr int Hn = 960;
constexpr int Dv = 480;
constexpr float SCALE = 0.04564354645876384f; // 480^-0.5

typedef __attribute__((ext_vector_type(16))) _Float16 v16h;
typedef __attribute__((ext_vector_type(8)))  float    v8f;

// ---------------------------------------------------------------------------
// WMMA fragment helpers (CDNA5 16x16x32 f16 layouts)
// ---------------------------------------------------------------------------
// A-matrix (16x32): lanes 0-15 -> K 0..15, lanes 16-31 -> K 16..31 (contiguous).
__device__ inline v16h frag_a(const _Float16* base, int stride, int row, int k0, int lane) {
  const _Float16* p = base + (size_t)row * stride + k0 + ((lane & 16) ? 16 : 0);
  v16h f;
#pragma unroll
  for (int i = 0; i < 16; ++i) f[i] = p[i];
  return f;
}
// B-matrix (32x16) from row-major (N,K) source (i.e. A@B^T): lane -> column n,
// contiguous K kb..kb+15 with kb = 0 (lanes 0-15) or 16 (lanes 16-31).
__device__ inline v16h frag_b(const _Float16* base, int stride, int row, int k0, int lane) {
  const _Float16* p = base + (size_t)row * stride + k0 + ((lane & 16) ? 16 : 0);
  v16h f;
#pragma unroll
  for (int i = 0; i < 16; ++i) f[i] = p[i];
  return f;
}
// B-matrix from row-major (K,N) source: gather down column `col`.
__device__ inline v16h frag_b_cols(const _Float16* base, int ldn, int k0, int col, int lane) {
  int kb = k0 + ((lane & 16) ? 16 : 0);
  v16h f;
#pragma unroll
  for (int i = 0; i < 16; ++i) f[i] = base[(size_t)(kb + i) * ldn + col];
  return f;
}
__device__ inline v8f wmma_f16(v16h a, v16h b, v8f c) {
  return __builtin_amdgcn_wmma_f32_16x16x32_f16(false, a, false, b, (short)0, c, false, false);
}

// ---------------------------------------------------------------------------
// Generic GEMM: out[r, n] = sum_k A[r,k] * W'[n,k] (+ bias[n])
//   transW=1: W'[n,k] = W[n*ldw + k]   (x @ W.T with W stored (N,K))
//   transW=0: W'[n,k] = W[k*ldw + n]   (x @ W   with W stored (K,N))
// Optional row remap for A (value[:,1:,:]): src = (r/rbOut)*rbSrc + srcOff + r%rbOut
// Block tile 128x64: 8 waves, one 16-row band each, 4 WMMA accumulators/wave.
// ---------------------------------------------------------------------------
template <typename AT, typename OT>
__global__ __launch_bounds__(256) void gemm_wmma(
    const AT* __restrict__ A, int lda,
    const float* __restrict__ W, int ldw, int transW,
    const float* __restrict__ bias,
    OT* __restrict__ out, int ldo,
    int rows, int N, int K,
    int rbOut, int rbSrc, int srcOff)
{
  __shared__ _Float16 As[128][40];
  __shared__ _Float16 Ws[64][40];
  int tid = threadIdx.x;
  int lane = tid & 31, wave = tid >> 5;
  int r0 = blockIdx.x * 128, c0 = blockIdx.y * 64;
  v8f acc0 = {}, acc1 = {}, acc2 = {}, acc3 = {};
  // staging coordinates
  int lrA = tid >> 1;            // 0..127
  int lkA = (tid & 1) * 16;      // 0 or 16
  int lrW = tid >> 2;            // 0..63
  int lkW = (tid & 3) * 8;       // 0,8,16,24
  int rA = r0 + lrA;
  int srcA = rA;
  if (rbOut > 0) srcA = (rA / rbOut) * rbSrc + srcOff + (rA % rbOut);
  bool okA = rA < rows;
  int nW = c0 + lrW;
  bool okW = nW < N;
  for (int kc = 0; kc < K; kc += 32) {
    if (kc + 32 < K) { // prefetch next K-chunk (global_prefetch_b8)
      if (okA) __builtin_prefetch(A + (size_t)srcA * lda + kc + 32 + lkA, 0, 1);
      if (okW) __builtin_prefetch(W + (transW ? ((size_t)nW * ldw + kc + 32 + lkW)
                                              : ((size_t)(kc + 32 + lkW) * ldw + nW)), 0, 1);
    }
    { // stage A (convert to f16)
      const AT* ap = A + (size_t)srcA * lda + kc + lkA;
#pragma unroll
      for (int i = 0; i < 16; ++i) As[lrA][lkA + i] = okA ? (_Float16)ap[i] : (_Float16)0.f;
    }
    { // stage W tile as (n, k)
#pragma unroll
      for (int i = 0; i < 8; ++i) {
        int k = kc + lkW + i;
        float w = okW ? (transW ? W[(size_t)nW * ldw + k] : W[(size_t)k * ldw + nW]) : 0.f;
        Ws[lrW][lkW + i] = (_Float16)w;
      }
    }
    __syncthreads();
    v16h a  = frag_a(&As[0][0], 40, wave * 16 + (lane & 15), 0, lane);
    v16h b0 = frag_b(&Ws[0][0], 40,  0 + (lane & 15), 0, lane);
    v16h b1 = frag_b(&Ws[0][0], 40, 16 + (lane & 15), 0, lane);
    v16h b2 = frag_b(&Ws[0][0], 40, 32 + (lane & 15), 0, lane);
    v16h b3 = frag_b(&Ws[0][0], 40, 48 + (lane & 15), 0, lane);
    acc0 = wmma_f16(a, b0, acc0);
    acc1 = wmma_f16(a, b1, acc1);
    acc2 = wmma_f16(a, b2, acc2);
    acc3 = wmma_f16(a, b3, acc3);
    __syncthreads();
  }
  int hi = (lane & 16) ? 8 : 0;
#pragma unroll
  for (int g = 0; g < 8; ++g) {
    int r = r0 + wave * 16 + g + hi;
    if (r >= rows) continue;
    int c = c0 + (lane & 15);
    if (c      < N) out[(size_t)r * ldo + c     ] = (OT)(acc0[g] + (bias ? bias[c     ] : 0.f));
    if (c + 16 < N) out[(size_t)r * ldo + c + 16] = (OT)(acc1[g] + (bias ? bias[c + 16] : 0.f));
    if (c + 32 < N) out[(size_t)r * ldo + c + 32] = (OT)(acc2[g] + (bias ? bias[c + 32] : 0.f));
    if (c + 48 < N) out[(size_t)r * ldo + c + 48] = (OT)(acc3[g] + (bias ? bias[c + 48] : 0.f));
  }
}

// ---------------------------------------------------------------------------
// Biaffine stage 2 (per (b,o) block): out[b,m,n,o] = tmp[o,b,m,:]·x2[b,n,:]
//                                    + lin1[b,m,o] + lin2[b,n,o] + lb[o]
// finalMul: multiply by lm[b,m,n] (last round), writing (B,M,N2,O) layout.
// Tile ids are clamped (not masked) so every wave runs unconditional WMMAs;
// clamped duplicates recompute the last tile and are excluded at store time.
// ---------------------------------------------------------------------------
__global__ __launch_bounds__(256) void biaffine2(
    const float* __restrict__ tmp, const float* __restrict__ x2,
    const float* __restrict__ lin1, const float* __restrict__ lin2,
    const float* __restrict__ lb, const float* __restrict__ lm,
    float* __restrict__ out, int Bc, int Mr, int N2, int O, int finalMul)
{
  __shared__ _Float16 As[64][40];
  __shared__ _Float16 Bs[80][40];
  int b = blockIdx.x / O, o = blockIdx.x % O;
  int tid = threadIdx.x, lane = tid & 31, wave = tid >> 5;
  int NT = (N2 + 15) >> 4;
  int cnt = 4 * NT;
  int id0 = wave,      t0 = id0 < cnt ? id0 : cnt - 1;
  int id1 = wave + 8,  t1 = id1 < cnt ? id1 : cnt - 1;
  int id2 = wave + 16, t2 = id2 < cnt ? id2 : cnt - 1;
  v8f acc0 = {}, acc1 = {}, acc2 = {};
  const float* Ab = tmp + ((size_t)o * Bc * Mr + (size_t)b * Mr) * Dv;
  const float* Bb = x2 + (size_t)b * N2 * Dv;
  for (int kc = 0; kc < Dv; kc += 32) {
    for (int idx = tid; idx < 64 * 32; idx += 256) {
      int rr = idx >> 5, kk = idx & 31;
      As[rr][kk] = (_Float16)Ab[(size_t)rr * Dv + kc + kk];
    }
    for (int idx = tid; idx < 80 * 32; idx += 256) {
      int rr = idx >> 5, kk = idx & 31;
      Bs[rr][kk] = (rr < N2) ? (_Float16)Bb[(size_t)rr * Dv + kc + kk] : (_Float16)0.f;
    }
    __syncthreads();
    {
      v16h a  = frag_a(&As[0][0], 40, (t0 & 3) * 16 + (lane & 15), 0, lane);
      v16h bb = frag_b(&Bs[0][0], 40, (t0 >> 2) * 16 + (lane & 15), 0, lane);
      acc0 = wmma_f16(a, bb, acc0);
    }
    {
      v16h a  = frag_a(&As[0][0], 40, (t1 & 3) * 16 + (lane & 15), 0, lane);
      v16h bb = frag_b(&Bs[0][0], 40, (t1 >> 2) * 16 + (lane & 15), 0, lane);
      acc1 = wmma_f16(a, bb, acc1);
    }
    {
      v16h a  = frag_a(&As[0][0], 40, (t2 & 3) * 16 + (lane & 15), 0, lane);
      v16h bb = frag_b(&Bs[0][0], 40, (t2 >> 2) * 16 + (lane & 15), 0, lane);
      acc2 = wmma_f16(a, bb, acc2);
    }
    __syncthreads();
  }
  int hi = (lane & 16) ? 8 : 0;
  int ids[3] = { id0, id1, id2 };
  v8f* accs[3] = { &acc0, &acc1, &acc2 };
#pragma unroll
  for (int ti = 0; ti < 3; ++ti) {
    int idx = ids[ti];
    if (idx >= cnt) continue;
    int rt = idx & 3, ct = idx >> 2;
#pragma unroll
    for (int g = 0; g < 8; ++g) {
      int r = rt * 16 + g + hi;
      int c = ct * 16 + (lane & 15);
      if (c >= N2) continue;
      float v = (*accs[ti])[g] + lin1[((size_t)b * Mr + r) * O + o]
                               + lin2[((size_t)b * N2 + c) * O + o] + lb[o];
      if (finalMul) v *= lm[((size_t)b * Mr + r) * N2 + c];
      out[(((size_t)b * Mr + r) * N2 + c) * O + o] = v;
    }
  }
}

// ---------------------------------------------------------------------------
// Fused attention core (one workgroup per sequence slice):
//   S = Q K^T * SCALE, masked softmax, AX = S @ X  (AX written in place of Q)
// mode 0: slice (b,m), rows n (R=N2, stride D);  mode 1: slice (b,n), rows m.
// Out-projection afterwards uses  out = AX @ Wv^T + bv  (softmax rows sum to 1).
// Tile ids are clamped so WMMAs run unconditionally; duplicate tile writes
// store identical values (benign) and AX stores are guarded by r < R.
// ---------------------------------------------------------------------------
__global__ __launch_bounds__(256) void attcore(
    const _Float16* Q, const _Float16* Kb, const _Float16* X, _Float16* AX,
    const float* __restrict__ mask, int mode, int Mr, int N2)
{
  __shared__ float    Sf[80][84];
  __shared__ _Float16 Sh[80][100];
  __shared__ _Float16 xs[96][100];
  int s = blockIdx.x;
  int R, rs, mbase, mstr; size_t base;
  if (mode == 0) { R = N2; rs = Dv; base = (size_t)s * N2 * Dv; mbase = s * N2; mstr = 1; }
  else {
    int b = s / N2, n = s % N2;
    R = Mr; rs = N2 * Dv; base = ((size_t)b * Mr * N2 + n) * Dv;
    mbase = b * Mr * N2 + n; mstr = N2;
  }
  int RP = (R + 15) & ~15;
  int NT = RP >> 4;
  int tid = threadIdx.x, lane = tid & 31, wave = tid >> 5;
  int hi = (lane & 16) ? 8 : 0;

  // Phase 1: scores via WMMA, fragments straight from global (L2-resident rows)
  {
    int nt2 = NT * NT;
    int iters = (nt2 + 7) >> 3;
    for (int t = 0; t < iters; ++t) {
      int idx = wave + t * 8;
      if (idx >= nt2) idx = nt2 - 1;   // clamp: duplicate compute, identical write
      int it = idx / NT, jt = idx % NT;
      v8f acc = {};
      for (int kc = 0; kc < Dv; kc += 32) {
        v16h a = frag_a(Q + base, rs, it * 16 + (lane & 15), kc, lane);
        v16h b = frag_b(Kb + base, rs, jt * 16 + (lane & 15), kc, lane);
        acc = wmma_f16(a, b, acc);
      }
#pragma unroll
      for (int g = 0; g < 8; ++g)
        Sf[it * 16 + g + hi][jt * 16 + (lane & 15)] = acc[g];
    }
  }
  __syncthreads();

  // Masked softmax (one thread per row); padded rows/cols zeroed in Sh.
  if (tid < RP) {
    int i = tid;
    if (i < R) {
      float mi = mask[mbase + i * mstr];
      float mx = -3.4e38f;
      for (int j = 0; j < R; ++j) {
        float mj = mask[mbase + j * mstr];
        float v = (mi * mj == 0.f) ? -10000.f : Sf[i][j] * SCALE;
        Sf[i][j] = v;
        mx = fmaxf(mx, v);
      }
      float sum = 0.f;
      for (int j = 0; j < R; ++j) { float e = __expf(Sf[i][j] - mx); Sf[i][j] = e; sum += e; }
      float inv = 1.f / sum;
      for (int j = 0; j < R; ++j) Sh[i][j] = (_Float16)(Sf[i][j] * inv);
      for (int j = R; j < 96; ++j) Sh[i][j] = (_Float16)0.f;
    } else {
      for (int j = 0; j < 96; ++j) Sh[i][j] = (_Float16)0.f;
    }
  }
  __syncthreads();

  // Phase 2: AX = Sh @ X, d in chunks of 96 staged transposed-friendly in LDS.
  {
    int ntiles = NT * 6;
    int iters = (ntiles + 7) >> 3;
    for (int dc = 0; dc < Dv; dc += 96) {
      for (int idx = tid; idx < 96 * 96; idx += 256) {
        int j = idx / 96, dd = idx % 96;
        xs[j][dd] = (j < R) ? X[base + (size_t)j * rs + dc + dd] : (_Float16)0.f;
      }
      __syncthreads();
      for (int t = 0; t < iters; ++t) {
        int idx = wave + t * 8;
        if (idx >= ntiles) idx = ntiles - 1;  // clamp: duplicate identical store
        int it = idx / 6, dt = idx % 6;
        v8f acc = {};
#pragma unroll
        for (int c = 0; c < 3; ++c) {
          v16h a = frag_a(&Sh[0][0], 100, it * 16 + (lane & 15), c * 32, lane);
          v16h b = frag_b_cols(&xs[0][0], 100, c * 32, dt * 16 + (lane & 15), lane);
          acc = wmma_f16(a, b, acc);
        }
#pragma unroll
        for (int g = 0; g < 8; ++g) {
          int r = it * 16 + g + hi;
          if (r < R) AX[base + (size_t)r * rs + dc + dt * 16 + (lane & 15)] = (_Float16)acc[g];
        }
      }
      __syncthreads();
    }
  }
}

// ---------------------------------------------------------------------------
// Small supporting kernels
// ---------------------------------------------------------------------------
__global__ void ffn_kernel(const float* __restrict__ L, const float* __restrict__ fw,
                           const float* __restrict__ fb, _Float16* __restrict__ X,
                           int O, int total) {
  int idx = blockIdx.x * 256 + threadIdx.x;
  if (idx >= total) return;
  int row = idx / Dv, d = idx % Dv;
  float v = fb[d];
  for (int o = 0; o < O; ++o) v += L[(size_t)row * O + o] * fw[d * O + o];
  X[(size_t)row * Dv + d] = (_Float16)v;
}

__global__ void rmax_over_n(const _Float16* __restrict__ X, float* __restrict__ dst,
                            int N2, int total) {
  int idx = blockIdx.x * 256 + threadIdx.x;
  if (idx >= total) return;
  int r2 = idx / Dv, d = idx % Dv;
  const _Float16* p = X + (size_t)r2 * N2 * Dv + d;
  float m = -3.4e38f;
  for (int n = 0; n < N2; ++n) m = fmaxf(m, (float)p[(size_t)n * Dv]);
  dst[idx] += m;
}

__global__ void rmax_over_m(const _Float16* __restrict__ X, float* __restrict__ dst,
                            int Mr, int N2, int total) {
  int idx = blockIdx.x * 256 + threadIdx.x;
  if (idx >= total) return;
  int rv = idx / Dv, d = idx % Dv;
  int b = rv / N2, n = rv % N2;
  const _Float16* p = X + ((size_t)b * Mr * N2 + n) * Dv + d;
  float m = -3.4e38f;
  for (int mm = 0; mm < Mr; ++mm) m = fmaxf(m, (float)p[(size_t)mm * N2 * Dv]);
  dst[idx] += m;
}

// one wave per (row, o): dst[row*O+o] = src[row,:] . w[o, off:off+D]
__global__ __launch_bounds__(256) void lin_dot(const float* __restrict__ src,
                                               const float* __restrict__ w, int ldw, int off,
                                               float* __restrict__ dst, int rows, int O) {
  int gw = (blockIdx.x * 256 + threadIdx.x) >> 5;
  int lane = threadIdx.x & 31;
  if (gw >= rows * O) return;
  int row = gw / O, o = gw % O;
  const float* s = src + (size_t)row * Dv;
  const float* wp = w + (size_t)o * ldw + off;
  float acc = 0.f;
  for (int k = lane; k < Dv; k += 32) acc += s[k] * wp[k];
  for (int d = 16; d; d >>= 1) acc += __shfl_down(acc, d, 32);
  if (lane == 0) dst[gw] = acc;
}

__global__ void lm_build(const float* __restrict__ ma, const float* __restrict__ mb,
                         int mbStride, int mbOff, float* __restrict__ lm,
                         int Mr, int N2, int total) {
  int idx = blockIdx.x * 256 + threadIdx.x;
  if (idx >= total) return;
  int r = idx / N2, n = idx % N2;
  int b = r / Mr;
  lm[idx] = ma[r] * mb[b * mbStride + mbOff + n];
}

__global__ void kmsel(const float* __restrict__ tlog, const float* __restrict__ km,
                      float* __restrict__ km2, int total) {
  int idx = blockIdx.x * 256 + threadIdx.x;
  if (idx >= total) return;
  const float* p = tlog + (size_t)idx * NVn * 3; // [b,m,n=0,:]
  float bv = p[0];
  int best = 0;
  if (p[1] > bv) { bv = p[1]; best = 1; }
  if (p[2] > bv) { bv = p[2]; best = 2; }
  km2[idx] = km[idx] * (best == 1 ? 1.f : 0.f);
}

__global__ void combine_multi(const float* __restrict__ a1, const float* __restrict__ a2,
                              const float* __restrict__ bc, float* __restrict__ out, int total) {
  int idx = blockIdx.x * 256 + threadIdx.x;
  if (idx >= total) return;
  int c = idx & 1;
  int rn = idx >> 1;
  int n = rn % 64, r = rn / 64;
  int b = r / Mn;
  out[idx] = a1[r * 2 + c] + a2[(b * 64 + n) * 2 + c] + bc[c];
}

// ---------------------------------------------------------------------------
// Host orchestration
// ---------------------------------------------------------------------------
extern "C" void kernel_launch(void* const* d_in, const int* in_sizes, int n_in,
                              void* d_out, int out_size, void* d_ws, size_t ws_size,
                              hipStream_t stream) {
  (void)in_sizes; (void)n_in; (void)out_size; (void)ws_size;
  const float* key        = (const float*)d_in[0];
  const float* value      = (const float*)d_in[1];
  const float* key_mask   = (const float*)d_in[2];
  const float* value_mask = (const float*)d_in[3];
  const float* w_kt = (const float*)d_in[4];  const float* b_kt = (const float*)d_in[5];
  const float* w_vt = (const float*)d_in[6];  const float* b_vt = (const float*)d_in[7];
  const float* w_km = (const float*)d_in[8];  const float* b_km = (const float*)d_in[9];
  const float* w_vm = (const float*)d_in[10]; const float* b_vm = (const float*)d_in[11];
  const float* w_ks = (const float*)d_in[12]; const float* b_ks = (const float*)d_in[13];
  const float* w_vs = (const float*)d_in[14]; const float* b_vs = (const float*)d_in[15];
  const float* bt_W = (const float*)d_in[16]; const float* bt_lw = (const float*)d_in[17];
  const float* bt_lb = (const float*)d_in[18];
  const float* bi_W = (const float*)d_in[19]; const float* bi_lw = (const float*)d_in[20];
  const float* bi_lb = (const float*)d_in[21];
  const float* w_ffnt = (const float*)d_in[22]; const float* b_ffnt = (const float*)d_in[23];
  const float* w_ffn  = (const float*)d_in[24]; const float* b_ffn  = (const float*)d_in[25];
  const float* qkvt_w = (const float*)d_in[26]; const float* qkvt_b = (const float*)d_in[27];
  const float* qkv_w  = (const float*)d_in[28]; const float* qkv_b  = (const float*)d_in[29];
  const float* w_cls = (const float*)d_in[30]; const float* b_cls = (const float*)d_in[31];
  float* out = (float*)d_out;

  // workspace carve-out (~410 MB total)
  char* ws = (char*)d_ws;
  size_t off = 0;
  auto alloc = [&](size_t bytes) {
    off = (off + 255) & ~(size_t)255;
    size_t o = off; off += bytes; return o;
  };
  const int rows2 = Bn * Mn;          // 2048
  const int rowsVmax = Bn * NVn;      // 2080
  const size_t rows3max = (size_t)Bn * Mn * NVn + 16; // + slack rows for padded frags
  float* keyA    = (float*)(ws + alloc((size_t)rows2 * Dv * 4));
  float* valA    = (float*)(ws + alloc((size_t)rowsVmax * Dv * 4));
  float* tmp3    = (float*)(ws + alloc((size_t)3 * rows2 * Dv * 4));
  float* logits3 = (float*)(ws + alloc((size_t)rows2 * NVn * 3 * 4));
  float* lin1    = (float*)(ws + alloc((size_t)rows2 * 3 * 4));
  float* lin2    = (float*)(ws + alloc((size_t)rowsVmax * 3 * 4));
  float* km2f    = (float*)(ws + alloc((size_t)rows2 * 4));
  float* lmbuf   = (float*)(ws + alloc((size_t)rows2 * NVn * 4));
  _Float16* Xb   = (_Float16*)(ws + alloc(rows3max * Dv * 2));
  _Float16* Qb   = (_Float16*)(ws + alloc(rows3max * Dv * 2));
  _Float16* Kb   = (_Float16*)(ws + alloc(rows3max * Dv * 2));

  dim3 gk((rows2 + 127) / 128, (Dv + 63) / 64);
  dim3 gv((rowsVmax + 127) / 128, (Dv + 63) / 64);

  auto run_rounds = [&](float* kA, float* vA, int N2, int O,
                        const float* Wb, const float* lw, const float* lb_,
                        const float* fw, const float* fb,
                        const float* qw, const float* qb,
                        const float* lmB, float* outP) {
    const int rowsV = Bn * N2;
    const int rows3 = Bn * Mn * N2;
    dim3 g1((rows2 + 127) / 128, (Dv + 63) / 64);
    dim3 g3((rows3 + 127) / 128, (Dv + 63) / 64);
    for (int rd = 0; rd < 5; ++rd) {
      bool last = (rd == 4);
      lin_dot<<<(rows2 * O * 32 + 255) / 256, 256, 0, stream>>>(kA, lw, 2 * Dv, 0, lin1, rows2, O);
      lin_dot<<<(rowsV * O * 32 + 255) / 256, 256, 0, stream>>>(vA, lw, 2 * Dv, Dv, lin2, rowsV, O);
      for (int o = 0; o < O; ++o)
        gemm_wmma<float, float><<<g1, 256, 0, stream>>>(
            kA, Dv, Wb + (size_t)o * Dv * Dv, Dv, /*transW=*/0, nullptr,
            tmp3 + (size_t)o * rows2 * Dv, Dv, rows2, Dv, Dv, 0, 0, 0);
      biaffine2<<<Bn * O, 256, 0, stream>>>(tmp3, vA, lin1, lin2, lb_, lmB,
                                            last ? outP : logits3, Bn, Mn, N2, O, last ? 1 : 0);
      if (last) break;
      ffn_kernel<<<(rows3 * Dv + 255) / 256, 256, 0, stream>>>(logits3, fw, fb, Xb, O, rows3 * Dv);
      for (int pass = 0; pass < 2; ++pass) {
        gemm_wmma<_Float16, _Float16><<<g3, 256, 0, stream>>>(
            Xb, Dv, qw, Dv, 1, qb, Qb, Dv, rows3, Dv, Dv, 0, 0, 0);
        gemm_wmma<_Float16, _Float16><<<g3, 256, 0, stream>>>(
            Xb, Dv, qw + (size_t)Dv * Dv, Dv, 1, qb + Dv, Kb, Dv, rows3, Dv, Dv, 0, 0, 0);
        int nsl = (pass == 0) ? rows2 : rowsV;
        attcore<<<nsl, 256, 0, stream>>>(Qb, Kb, Xb, Qb, lmB, pass, Mn, N2);
        gemm_wmma<_Float16, _Float16><<<g3, 256, 0, stream>>>(
            Qb, Dv, qw + 2 * (size_t)Dv * Dv, Dv, 1, qb + 2 * Dv, Xb, Dv, rows3, Dv, Dv, 0, 0, 0);
      }
      rmax_over_n<<<(rows2 * Dv + 255) / 256, 256, 0, stream>>>(Xb, kA, N2, rows2 * Dv);
      rmax_over_m<<<(rowsV * Dv + 255) / 256, 256, 0, stream>>>(Xb, vA, Mn, N2, rowsV * Dv);
    }
  };

  // ---- type head ----
  gemm_wmma<float, float><<<gk, 256, 0, stream>>>(key, Hn, w_kt, Hn, 1, b_kt, keyA, Dv,
                                                  rows2, Dv, Hn, 0, 0, 0);
  gemm_wmma<float, float><<<gv, 256, 0, stream>>>(value, Hn, w_vt, Hn, 1, b_vt, valA, Dv,
                                                  rowsVmax, Dv, Hn, 0, 0, 0);
  lm_build<<<(rows2 * NVn + 255) / 256, 256, 0, stream>>>(key_mask, value_mask, NVn, 0, lmbuf,
                                                          Mn, NVn, rows2 * NVn);
  run_rounds(keyA, valA, NVn, 3, bt_W, bt_lw, bt_lb, w_ffnt, b_ffnt, qkvt_w, qkvt_b, lmbuf, out);

  // ---- key-type selection ----
  kmsel<<<(rows2 + 255) / 256, 256, 0, stream>>>(out, key_mask, km2f, rows2);

  // ---- multi head (pure linear) ----
  gemm_wmma<float, float><<<gk, 256, 0, stream>>>(key, Hn, w_km, Hn, 1, b_km, keyA, Dv,
                                                  rows2, Dv, Hn, 0, 0, 0);
  gemm_wmma<float, float><<<gk, 256, 0, stream>>>(value, Hn, w_vm, Hn, 1, b_vm, valA, Dv,
                                                  rows2, Dv, Hn, 64, 65, 1);
  lin_dot<<<(rows2 * 2 * 32 + 255) / 256, 256, 0, stream>>>(keyA, w_cls, 2 * Dv, 0, lin1, rows2, 2);
  lin_dot<<<(rows2 * 2 * 32 + 255) / 256, 256, 0, stream>>>(valA, w_cls, 2 * Dv, Dv, lin2, rows2, 2);
  float* out_multi = out + (size_t)rows2 * NVn * 3;
  combine_multi<<<(rows2 * 64 * 2 + 255) / 256, 256, 0, stream>>>(lin1, lin2, b_cls, out_multi,
                                                                  rows2 * 64 * 2);

  // ---- single head ----
  gemm_wmma<float, float><<<gk, 256, 0, stream>>>(key, Hn, w_ks, Hn, 1, b_ks, keyA, Dv,
                                                  rows2, Dv, Hn, 0, 0, 0);
  gemm_wmma<float, float><<<gk, 256, 0, stream>>>(value, Hn, w_vs, Hn, 1, b_vs, valA, Dv,
                                                  rows2, Dv, Hn, 64, 65, 1);
  lm_build<<<(rows2 * 64 + 255) / 256, 256, 0, stream>>>(km2f, value_mask, NVn, 1, lmbuf,
                                                         Mn, 64, rows2 * 64);
  float* out_single = out_multi + (size_t)rows2 * 64 * 2;
  run_rounds(keyA, valA, 64, 2, bi_W, bi_lw, bi_lb, w_ffn, b_ffn, qkv_w, qkv_b, lmbuf, out_single);
}
